// BREWAAttention_84293028151343
// MI455X (gfx1250) — compile-verified, matrix operand order checked by reading, not run on hardware
//
#include <hip/hip_runtime.h>
#include <hip/hip_bf16.h>
#include <math.h>

// ---------------------------------------------------------------------------
// BREWA attention for MI455X (gfx1250, wave32, WMMA 16x16x32 f16 -> f32 acc)
// ---------------------------------------------------------------------------

typedef _Float16 v8h  __attribute__((ext_vector_type(8)));
typedef _Float16 v16h __attribute__((ext_vector_type(16)));
typedef float    v8f  __attribute__((ext_vector_type(8)));
typedef int      v4i  __attribute__((vector_size(16)));

constexpr int kB   = 4;
constexpr int kN   = 2048;
constexpr int kD   = 1024;
constexpr int kH   = 16;
constexpr int kHd  = 64;
constexpr int kMb  = 32;
constexpr int kM   = kB * kN;        // 8192 total rows
constexpr int kEnc = kH * kMb;       // 512 enc cols per row

// Async global->LDS (ASYNCcnt) path, when the toolchain declares the builtins.
#if defined(__gfx1250__) && __has_builtin(__builtin_amdgcn_global_load_async_to_lds_b128) && __has_builtin(__builtin_amdgcn_s_wait_asynccnt)
#define USE_ASYNC_LDS 1
typedef __attribute__((address_space(1))) v4i* gvec_ptr;
typedef __attribute__((address_space(3))) v4i* lvec_ptr;
#endif

__device__ __forceinline__ void copy16B_g2l(const _Float16* g, _Float16* l) {
#if USE_ASYNC_LDS
  __builtin_amdgcn_global_load_async_to_lds_b128((gvec_ptr)g, (lvec_ptr)l, 0, 0);
#else
  *(v8h*)l = *(const v8h*)g;
#endif
}

__device__ __forceinline__ void wait_async_all() {
#if USE_ASYNC_LDS
  __builtin_amdgcn_s_wait_asynccnt(0);
#endif
}

__device__ __forceinline__ v8f vzero8() {
  v8f z;
#pragma unroll
  for (int i = 0; i < 8; ++i) z[i] = 0.0f;
  return z;
}

// Native V_TANH_F32 when exposed; branch-free __expf fallback otherwise.
__device__ __forceinline__ float fast_tanh(float x) {
#if defined(__gfx1250__) && __has_builtin(__builtin_amdgcn_tanhf)
  return __builtin_amdgcn_tanhf(x);
#else
  x = fminf(fmaxf(x, -10.0f), 10.0f);
  const float e = __expf(2.0f * x);
  return (e - 1.0f) / (e + 1.0f);
#endif
}

// One WMMA: D = A(16x32 f16) * B(32x16 f16) + C(16x16 f32)
__device__ __forceinline__ v8f wmma16x16x32(v16h a, v16h b, v8f c) {
  return __builtin_amdgcn_wmma_f32_16x16x32_f16(
      /*neg_a=*/false, a, /*neg_b=*/false, b,
      /*c_mod=*/(short)0, c, /*reuse_a=*/false, /*reuse_b=*/false);
}

// Load a 16x32 f16 fragment whose "row" dimension has stride `ld` halves and
// whose K dimension is contiguous. Used for BOTH A fragments (row = M) and
// B fragments of X*W^T-style products (row = N, since W is stored [N][K]).
// Per ISA 7.12.2 (16-bit A 16x32): lane L<16 -> row L, K = {0..7, 16..23};
// lane L>=16 -> row L-16, K = {8..15, 24..31}. Two 16B loads per lane.
__device__ __forceinline__ v16h load_frag(const _Float16* __restrict__ p0, int ld) {
  const int lane = threadIdx.x & 31;
  const _Float16* p = p0 + (size_t)(lane & 15) * (size_t)ld + ((lane >> 4) << 3);
  v8h lo = *(const v8h*)(p);       // K = kb .. kb+7
  v8h hi = *(const v8h*)(p + 16);  // K = 16+kb .. 16+kb+7
  v16h f;
#pragma unroll
  for (int j = 0; j < 8; ++j) { f[j] = lo[j]; f[j + 8] = hi[j]; }
  return f;
}

// ---------------------------------------------------------------------------
// 1) f32 -> f16 conversion
// ---------------------------------------------------------------------------
__global__ void cvt_f16_kernel(const float* __restrict__ s,
                               _Float16* __restrict__ d, int n) {
  int i = (blockIdx.x * blockDim.x + threadIdx.x) * 4;
  if (i + 3 < n) {
    float4 f = *(const float4*)(s + i);
    d[i + 0] = (_Float16)f.x;
    d[i + 1] = (_Float16)f.y;
    d[i + 2] = (_Float16)f.z;
    d[i + 3] = (_Float16)f.w;
  } else {
    for (; i < n; ++i) d[i] = (_Float16)s[i];
  }
}

// ---------------------------------------------------------------------------
// 2) GEMM: Y[M,Nn] = X[M,K] * W[Nn,K]^T   (both operands K-contiguous)
//    128x128 macro tile, BK=32, 8 waves: each wave owns 32x64 (8 WMMA tiles).
//    Double-buffered LDS; next k-tile staged with GLOBAL_LOAD_ASYNC_TO_LDS
//    while WMMAs run on the current buffer.
// ---------------------------------------------------------------------------
template <typename OutT>
__global__ void __launch_bounds__(256)
gemm_nt_kernel(const _Float16* __restrict__ X,
               const _Float16* __restrict__ W,
               OutT* __restrict__ Y,
               int M, int Nn, int K) {
  __shared__ _Float16 sA[2][128][40];  // [buf][m][k], 16B-aligned rows
  __shared__ _Float16 sB[2][128][40];  // [buf][n][k]
  const int tid  = threadIdx.x;
  const int lane = tid & 31;
  const int wave = tid >> 5;
  const int m0 = blockIdx.y * 128;
  const int n0 = blockIdx.x * 128;
  const int wm = (wave & 3) * 32;   // wave's row offset in tile
  const int wn = (wave >> 2) * 64;  // wave's col offset in tile

  v8f acc[2][4];
#pragma unroll
  for (int i = 0; i < 2; ++i)
#pragma unroll
    for (int j = 0; j < 4; ++j) acc[i][j] = vzero8();

  const int ldr = tid >> 1;          // 0..127 : tile row this thread stages
  const int seg = (tid & 1) * 16;    // 0 or 16: which 16-half segment

  auto stage = [&](int buf, int kt) {
    const _Float16* gx = X + (size_t)(m0 + ldr) * K + kt + seg;
    const _Float16* gw = W + (size_t)(n0 + ldr) * K + kt + seg;
    copy16B_g2l(gx,     &sA[buf][ldr][seg]);
    copy16B_g2l(gx + 8, &sA[buf][ldr][seg + 8]);
    copy16B_g2l(gw,     &sB[buf][ldr][seg]);
    copy16B_g2l(gw + 8, &sB[buf][ldr][seg + 8]);
  };

  stage(0, 0);
  wait_async_all();
  __syncthreads();

  int buf = 0;
  for (int kt = 0; kt < K; kt += 32) {
    if (kt + 32 < K) stage(buf ^ 1, kt + 32);  // overlap with WMMAs below

    v16h a[2], b[4];
    a[0] = load_frag(&sA[buf][wm][0], 40);
    a[1] = load_frag(&sA[buf][wm + 16][0], 40);
#pragma unroll
    for (int j = 0; j < 4; ++j) b[j] = load_frag(&sB[buf][wn + j * 16][0], 40);
#pragma unroll
    for (int i = 0; i < 2; ++i)
#pragma unroll
      for (int j = 0; j < 4; ++j)
        acc[i][j] = wmma16x16x32(a[i], b[j], acc[i][j]);

    wait_async_all();
    __syncthreads();
    buf ^= 1;
  }

  // C/D layout: VGPR r -> row r (lanes 0-15) / r+8 (lanes 16-31); col = lane&15
  const int cn = lane & 15;
  const int ro = (lane >> 4) << 3;
#pragma unroll
  for (int i = 0; i < 2; ++i)
#pragma unroll
    for (int j = 0; j < 4; ++j) {
      const int col = n0 + wn + j * 16 + cn;
      const int row = m0 + wm + i * 16 + ro;
#pragma unroll
      for (int r = 0; r < 8; ++r)
        Y[(size_t)(row + r) * Nn + col] = (OutT)acc[i][j][r];
    }
}

// ---------------------------------------------------------------------------
// 3) Witness encoder: Enc[bn, h*32+m] = tanh( sum_d Qh[bn, h*64+d] * Wenc[h,d,m] )
//    Per block: one head h, 128 rows; 8 waves x 16 rows; K=64 -> 2 WMMA steps.
// ---------------------------------------------------------------------------
__global__ void __launch_bounds__(256)
enc_kernel(const _Float16* __restrict__ Qh,    // [kM][kD]
           const _Float16* __restrict__ Wenc,  // [kH][kHd][kMb]
           _Float16* __restrict__ Enc) {       // [kM][kEnc]
  __shared__ _Float16 sW[kMb][72];             // transposed: [m][d]
  const int tid = threadIdx.x, lane = tid & 31, wave = tid >> 5;
  const int h  = blockIdx.y;
  const int m0 = blockIdx.x * 128;

  {  // stage Wenc[h] transposed into LDS
    const int d  = tid >> 2;         // 0..63
    const int mm = (tid & 3) * 8;    // 0,8,16,24
    v8h v = *(const v8h*)(Wenc + ((size_t)h * kHd + d) * kMb + mm);
#pragma unroll
    for (int j = 0; j < 8; ++j) sW[mm + j][d] = v[j];
  }
  __syncthreads();

  const int qrow = m0 + wave * 16;
  v8f acc[2] = {vzero8(), vzero8()};
#pragma unroll
  for (int ks = 0; ks < 2; ++ks) {
    v16h a = load_frag(Qh + (size_t)qrow * kD + h * kHd + ks * 32, kD);
#pragma unroll
    for (int j = 0; j < 2; ++j) {
      v16h bf = load_frag(&sW[j * 16][ks * 32], 72);
      acc[j] = wmma16x16x32(a, bf, acc[j]);
    }
  }
  const int cn = lane & 15, ro = (lane >> 4) << 3;
#pragma unroll
  for (int j = 0; j < 2; ++j) {
    const int col = h * kMb + j * 16 + cn;
#pragma unroll
    for (int r = 0; r < 8; ++r)
      Enc[(size_t)(qrow + ro + r) * kEnc + col] = (_Float16)fast_tanh(acc[j][r]);
  }
}

// ---------------------------------------------------------------------------
// 4) Flash attention per (b,h): S = Qe*Ke^T/sqrt(32), online softmax, O = P*V
//    Block = 128 query rows; wave owns 16 rows; key blocks of 32.
// ---------------------------------------------------------------------------
__global__ void __launch_bounds__(256)
attn_kernel(const _Float16* __restrict__ Qe,  // [kM][kEnc]
            const _Float16* __restrict__ Ke,  // [kM][kEnc]
            const _Float16* __restrict__ V,   // [kM][kD]
            _Float16* __restrict__ Ao) {      // [kM][kD]
  __shared__ _Float16 sV[64][40];        // [dim][key] (transposed V block)
  __shared__ _Float16 sP[8][16][40];     // per-wave probabilities
  __shared__ float    sS[8][16][36];     // per-wave raw scores
  __shared__ float    sAl[8][16];        // per-row rescale alpha
  __shared__ float    sSum[8][16];       // per-row final softmax denom

  const int tid = threadIdx.x, lane = tid & 31, wave = tid >> 5;
  const int bh = blockIdx.y;
  const int bb = bh >> 4;
  const int h  = bh & 15;
  const int q0 = blockIdx.x * 128 + wave * 16;
  const size_t rowQ = (size_t)bb * kN + q0;
  const float scale = 0.17677669529663688f;  // 1/sqrt(32)

  const v16h aQ = load_frag(Qe + rowQ * kEnc + h * kMb, kEnc);

  v8f acc[4];
#pragma unroll
  for (int j = 0; j < 4; ++j) acc[j] = vzero8();
  float mrun = -1e30f, lrun = 0.0f;

  const int cn = lane & 15;
  const int ro = (lane >> 4) << 3;

  for (int kb = 0; kb < kN; kb += 32) {
    {  // stage V[kb..kb+31][h*64..h*64+63] transposed into sV[dim][key]
      const int key = tid >> 3;        // 0..31
      const int dd  = (tid & 7) * 8;   // 0..56
      v8h v = *(const v8h*)(V + ((size_t)bb * kN + kb + key) * kD + h * kHd + dd);
#pragma unroll
      for (int j = 0; j < 8; ++j) sV[dd + j][key] = v[j];
    }
    // score WMMAs straight from global (k_enc is m-contiguous)
    {
      const _Float16* kbase = Ke + ((size_t)bb * kN + kb) * kEnc + h * kMb;
      v16h b0 = load_frag(kbase, kEnc);
      v16h b1 = load_frag(kbase + (size_t)16 * kEnc, kEnc);
      v8f s0 = wmma16x16x32(aQ, b0, vzero8());
      v8f s1 = wmma16x16x32(aQ, b1, vzero8());
#pragma unroll
      for (int r = 0; r < 8; ++r) {
        sS[wave][ro + r][cn]      = s0[r] * scale;
        sS[wave][ro + r][16 + cn] = s1[r] * scale;
      }
    }
    __syncthreads();

    if (lane < 16) {  // online softmax, lane l owns query row l of this wave
      float smax = mrun;
#pragma unroll
      for (int k = 0; k < 32; ++k) smax = fmaxf(smax, sS[wave][lane][k]);
      const float alpha = __expf(mrun - smax);
      float ls = 0.0f;
#pragma unroll
      for (int k = 0; k < 32; ++k) {
        const float p = __expf(sS[wave][lane][k] - smax);
        sP[wave][lane][k] = (_Float16)p;
        ls += p;
      }
      lrun = lrun * alpha + ls;
      mrun = smax;
      sAl[wave][lane] = alpha;
    }
    __syncthreads();

    // rescale accumulators, then O += P * V
#pragma unroll
    for (int r = 0; r < 8; ++r) {
      const float al = sAl[wave][ro + r];
#pragma unroll
      for (int j = 0; j < 4; ++j) acc[j][r] *= al;
    }
    const v16h aP = load_frag(&sP[wave][0][0], 40);
#pragma unroll
    for (int j = 0; j < 4; ++j) {
      v16h bV = load_frag(&sV[j * 16][0], 40);
      acc[j] = wmma16x16x32(aP, bV, acc[j]);
    }
    __syncthreads();
  }

  if (lane < 16) sSum[wave][lane] = lrun;
  __syncthreads();

#pragma unroll
  for (int j = 0; j < 4; ++j) {
#pragma unroll
    for (int r = 0; r < 8; ++r) {
      const float o = acc[j][r] / sSum[wave][ro + r];
      Ao[(rowQ + ro + r) * kD + h * kHd + j * 16 + cn] = (_Float16)o;
    }
  }
}

// ---------------------------------------------------------------------------
// Host-side launch
// ---------------------------------------------------------------------------
extern "C" void kernel_launch(void* const* d_in, const int* in_sizes, int n_in,
                              void* d_out, int out_size, void* d_ws, size_t ws_size,
                              hipStream_t stream) {
  const float* x    = (const float*)d_in[0];
  const float* Wq   = (const float*)d_in[1];
  const float* Wk   = (const float*)d_in[2];
  const float* Wv   = (const float*)d_in[3];
  const float* Wenc = (const float*)d_in[4];
  const float* Wo   = (const float*)d_in[5];

  // workspace carve-up (f16 elements); total ~110 MB
  _Float16* p = (_Float16*)d_ws;
  auto take = [&](size_t n) { _Float16* r = p; p += n; return r; };
  const size_t MD = (size_t)kM * kD;
  const size_t DD = (size_t)kD * kD;
  _Float16* hX    = take(MD);
  _Float16* hWq   = take(DD);
  _Float16* hWk   = take(DD);
  _Float16* hWv   = take(DD);
  _Float16* hWo   = take(DD);
  _Float16* hWenc = take((size_t)kH * kHd * kMb);
  _Float16* hQ    = take(MD);
  _Float16* hK    = take(MD);
  _Float16* hV    = take(MD);
  _Float16* hQe   = take((size_t)kM * kEnc);
  _Float16* hKe   = take((size_t)kM * kEnc);
  _Float16* hAo   = take(MD);

  auto cvt = [&](const float* s, _Float16* d, size_t n) {
    const int thr = 256;
    const int blocks = (int)((n / 4 + thr - 1) / thr);
    cvt_f16_kernel<<<blocks, thr, 0, stream>>>(s, d, (int)n);
  };
  cvt(x, hX, MD);
  cvt(Wq, hWq, DD);
  cvt(Wk, hWk, DD);
  cvt(Wv, hWv, DD);
  cvt(Wo, hWo, DD);
  cvt(Wenc, hWenc, (size_t)kH * kHd * kMb);

  // QKV projections: Y = X * W^T
  dim3 gg(kD / 128, kM / 128);
  gemm_nt_kernel<_Float16><<<gg, 256, 0, stream>>>(hX, hWq, hQ, kM, kD, kD);
  gemm_nt_kernel<_Float16><<<gg, 256, 0, stream>>>(hX, hWk, hK, kM, kD, kD);
  gemm_nt_kernel<_Float16><<<gg, 256, 0, stream>>>(hX, hWv, hV, kM, kD, kD);

  // Witness encoders
  dim3 ge(kM / 128, kH);
  enc_kernel<<<ge, 256, 0, stream>>>(hQ, hWenc, hQe);
  enc_kernel<<<ge, 256, 0, stream>>>(hK, hWenc, hKe);

  // Flash attention over witness scores
  dim3 ga(kN / 128, kB * kH);
  attn_kernel<<<ga, 256, 0, stream>>>(hQe, hKe, hV, hAo);

  // Output projection (f32 out)
  gemm_nt_kernel<float><<<gg, 256, 0, stream>>>(hAo, hWo, (float*)d_out, kM, kD, kD);
}
